// PointMamba_48077863911652
// MI455X (gfx1250) — compile-verified, matrix operand order checked by reading, not run on hardware
//
#include <hip/hip_runtime.h>
#include <hip/hip_bf16.h>
#include <math.h>

// ---------------- model constants ----------------
static const int Bc = 8;      // batch
static const int Nc = 2048;   // points
static const int Pc = 64;     // patches/centers
static const int Kc = 32;     // knn
static const int Dc = 384;    // token dim
static const int Lc = 128;    // sequence length (2*P)
static const int DIc = 768;
static const int DSc = 16;
static const int DCONVc = 4;
static const int DTRc = 24;
static const int NCc = 40;
static const int DEPTHc = 12;

typedef __attribute__((ext_vector_type(16))) _Float16 v16h;
typedef __attribute__((ext_vector_type(8)))  float    v8f;
typedef __attribute__((ext_vector_type(4)))  float    f4;
typedef __attribute__((ext_vector_type(4)))  _Float16 h4;

__device__ __forceinline__ float gelu_exact(float x) {
    return 0.5f * x * (1.0f + erff(x * 0.70710678118654752f));
}
__device__ __forceinline__ float silu_f(float x) {
    return x / (1.0f + __expf(-x));
}

// ======================================================================
// KNN: one block per (b,p); compute d2 to all N points, iteratively pick
// 32 minima (tie-break by smallest index, matching stable top_k).
// ======================================================================
__global__ __launch_bounds__(256) void knn_kernel(const float* __restrict__ xyz,
                                                  float* __restrict__ centers,
                                                  int* __restrict__ knn) {
    int b = blockIdx.x / Pc, p = blockIdx.x % Pc;
    const float* X = xyz + (size_t)b * Nc * 3;
    __shared__ float d2[Nc];
    __shared__ float cx[3];
    __shared__ float rv[256];
    __shared__ int   ri[256];
    int tid = threadIdx.x;
    if (tid < 3) {
        float v = X[(p * (Nc / Pc)) * 3 + tid];
        cx[tid] = v;
        centers[((size_t)b * Pc + p) * 3 + tid] = v;
    }
    __syncthreads();
    float c0 = cx[0], c1 = cx[1], c2 = cx[2];
    for (int i = tid; i < Nc; i += 256) {
        float dx = X[i * 3 + 0] - c0;
        float dy = X[i * 3 + 1] - c1;
        float dz = X[i * 3 + 2] - c2;
        d2[i] = dx * dx + dy * dy + dz * dz;
    }
    __syncthreads();
    for (int kk = 0; kk < Kc; ++kk) {
        float bv = 3.4e38f; int bi = 0x7fffffff;
        for (int i = tid; i < Nc; i += 256) {
            float v = d2[i];
            if (v < bv || (v == bv && i < bi)) { bv = v; bi = i; }
        }
        rv[tid] = bv; ri[tid] = bi;
        __syncthreads();
        for (int o = 128; o > 0; o >>= 1) {
            if (tid < o) {
                float v2 = rv[tid + o]; int i2 = ri[tid + o];
                if (v2 < rv[tid] || (v2 == rv[tid] && i2 < ri[tid])) { rv[tid] = v2; ri[tid] = i2; }
            }
            __syncthreads();
        }
        if (tid == 0) {
            knn[((size_t)b * Pc + p) * Kc + kk] = ri[0];
            d2[ri[0]] = 3.4e38f;
        }
        __syncthreads();
    }
}

// ======================================================================
// Patch MLP: one block per (b,p). 3->64->128->384 with BN(eval)+GELU,
// maxpool over K neighbors.
// ======================================================================
__global__ __launch_bounds__(256) void patch_mlp_kernel(
    const float* __restrict__ xyz, const float* __restrict__ centers,
    const int* __restrict__ knn,
    const float* __restrict__ w1, const float* __restrict__ b1,
    const float* __restrict__ g1, const float* __restrict__ be1,
    const float* __restrict__ w2, const float* __restrict__ b2,
    const float* __restrict__ g2, const float* __restrict__ be2,
    const float* __restrict__ w3, const float* __restrict__ b3,
    const float* __restrict__ g3, const float* __restrict__ be3,
    float* __restrict__ tokens) {
    int b = blockIdx.x / Pc, p = blockIdx.x % Pc;
    __shared__ float nx[Kc][3];
    __shared__ float h1[Kc][64];
    __shared__ float h2[Kc][128];
    const int*   kn = knn + ((size_t)b * Pc + p) * Kc;
    const float* X  = xyz + (size_t)b * Nc * 3;
    const float* C  = centers + ((size_t)b * Pc + p) * 3;
    const float  bns = rsqrtf(1.0f + 1e-5f);
    int tid = threadIdx.x;
    for (int i = tid; i < Kc * 3; i += blockDim.x) {
        int k = i / 3, d = i % 3;
        nx[k][d] = X[kn[k] * 3 + d] - C[d];
    }
    __syncthreads();
    for (int i = tid; i < Kc * 64; i += blockDim.x) {
        int k = i >> 6, o = i & 63;
        float s = b1[o] + nx[k][0] * w1[o * 3 + 0] + nx[k][1] * w1[o * 3 + 1] + nx[k][2] * w1[o * 3 + 2];
        s = s * (g1[o] * bns) + be1[o];
        h1[k][o] = gelu_exact(s);
    }
    __syncthreads();
    for (int i = tid; i < Kc * 128; i += blockDim.x) {
        int k = i >> 7, o = i & 127;
        float s = b2[o];
        const float* wr = w2 + (size_t)o * 64;
        #pragma unroll 8
        for (int j = 0; j < 64; ++j) s += h1[k][j] * wr[j];
        s = s * (g2[o] * bns) + be2[o];
        h2[k][o] = gelu_exact(s);
    }
    __syncthreads();
    for (int o = tid; o < Dc; o += blockDim.x) {
        const float* wr = w3 + (size_t)o * 128;
        float mx = -3.4e38f;
        for (int k = 0; k < Kc; ++k) {
            float s = b3[o];
            #pragma unroll 8
            for (int j = 0; j < 128; ++j) s += h2[k][j] * wr[j];
            s = s * (g3[o] * bns) + be3[o];
            mx = fmaxf(mx, s);
        }
        tokens[((size_t)b * Pc + p) * Dc + o] = mx;
    }
}

// ======================================================================
// Z-order (Morton) sort of centers: one block per batch, 64 threads.
// ======================================================================
__device__ __forceinline__ unsigned spread_bits_d(unsigned v) {
    v &= 1023u;
    v = (v | (v << 16)) & 50331903u;
    v = (v | (v << 8))  & 50393103u;
    v = (v | (v << 4))  & 51130563u;
    v = (v | (v << 2))  & 153391689u;
    return v;
}

__global__ __launch_bounds__(64) void sfc_kernel(const float* __restrict__ centers,
                                                 int* __restrict__ oh, int* __restrict__ oth) {
    int b = blockIdx.x;
    int t = threadIdx.x;
    __shared__ float cs[Pc][3];
    __shared__ unsigned kh[Pc], kt[Pc];
    __shared__ float lo[3], hi[3];
    for (int d = 0; d < 3; ++d) cs[t][d] = centers[((size_t)b * Pc + t) * 3 + d];
    __syncthreads();
    if (t < 3) {
        float mn = cs[0][t], mx = cs[0][t];
        for (int i = 1; i < Pc; ++i) { mn = fminf(mn, cs[i][t]); mx = fmaxf(mx, cs[i][t]); }
        lo[t] = mn; hi[t] = mx;
    }
    __syncthreads();
    int q[3];
    for (int d = 0; d < 3; ++d) {
        float f = (cs[t][d] - lo[d]) / (hi[d] - lo[d] + 1e-6f) * 1023.0f;
        int qi = (int)f;
        q[d] = min(max(qi, 0), 1023);
    }
    kh[t] = spread_bits_d((unsigned)q[0]) | (spread_bits_d((unsigned)q[1]) << 1) | (spread_bits_d((unsigned)q[2]) << 2);
    kt[t] = spread_bits_d((unsigned)q[2]) | (spread_bits_d((unsigned)q[1]) << 1) | (spread_bits_d((unsigned)q[0]) << 2);
    __syncthreads();
    int rh = 0, rt = 0;
    unsigned mh = kh[t], mt = kt[t];
    for (int j = 0; j < Pc; ++j) {
        unsigned kj = kh[j]; if (kj < mh || (kj == mh && j < t)) rh++;
        unsigned k2 = kt[j]; if (k2 < mt || (k2 == mt && j < t)) rt++;
    }
    oh[b * Pc + rh]  = t;
    oth[b * Pc + rt] = t;
}

// ======================================================================
// Build token sequence t[b, 0:128, :] with reorder + scale/shift + pos.
// ======================================================================
__global__ void build_t_kernel(const float* __restrict__ tokens,
                               const int* __restrict__ oh, const int* __restrict__ oth,
                               const float* __restrict__ hs, const float* __restrict__ hh,
                               const float* __restrict__ ts, const float* __restrict__ th,
                               const float* __restrict__ pos, float* __restrict__ t) {
    int idx = blockIdx.x * blockDim.x + threadIdx.x;
    int total = Bc * Lc * Dc;
    if (idx >= total) return;
    int d = idx % Dc;
    int i = (idx / Dc) % Lc;
    int b = idx / (Dc * Lc);
    float v;
    if (i < Pc) {
        int src = oh[b * Pc + i];
        v = tokens[((size_t)b * Pc + src) * Dc + d] * hs[d] + hh[d] + pos[i * Dc + d];
    } else {
        int src = oth[b * Pc + (i - Pc)];
        v = tokens[((size_t)b * Pc + src) * Dc + d] * ts[d] + th[d] + pos[(i - Pc) * Dc + d];
    }
    t[idx] = v;
}

// ======================================================================
// LayerNorm over last dim: one block (128 threads) per row.
// ======================================================================
__global__ __launch_bounds__(128) void ln_kernel(const float* __restrict__ x,
                                                 const float* __restrict__ g,
                                                 const float* __restrict__ b,
                                                 float* __restrict__ y, int Dd) {
    int row = blockIdx.x;
    const float* xr = x + (size_t)row * Dd;
    __shared__ float red[128];
    int tid = threadIdx.x;
    float s = 0.f;
    for (int i = tid; i < Dd; i += 128) s += xr[i];
    red[tid] = s; __syncthreads();
    for (int o = 64; o > 0; o >>= 1) { if (tid < o) red[tid] += red[tid + o]; __syncthreads(); }
    float mean = red[0] / (float)Dd;
    __syncthreads();
    float v = 0.f;
    for (int i = tid; i < Dd; i += 128) { float d = xr[i] - mean; v += d * d; }
    red[tid] = v; __syncthreads();
    for (int o = 64; o > 0; o >>= 1) { if (tid < o) red[tid] += red[tid + o]; __syncthreads(); }
    float inv = rsqrtf(red[0] / (float)Dd + 1e-5f);
    for (int i = tid; i < Dd; i += 128)
        y[(size_t)row * Dd + i] = (xr[i] - mean) * inv * g[i] + b[i];
}

// ======================================================================
// Generic WMMA GEMM: C[M,N] = A[M,K] * W[N,K]^T (+bias)(+softplus)(+Cadd)
// Block tile 128x64, 8 waves, each wave 32x32 via 2x2 v_wmma_f32_16x16x32_f16.
// fp32 inputs staged as float4 global loads -> f16 -> 8B LDS vector stores,
// software-pipelined so the next tile's loads issue before this tile's WMMAs.
// Requires Kdim % 4 == 0 (vector-granular bounds checks).
// ======================================================================
#define GBM 128
#define GBN 64
#define GBK 32
#define GLDST 36   // halves: 72B row stride -> 8B-aligned vec stores, 18-bank stride

__global__ __launch_bounds__(256) void gemm_wmma_kernel(
    const float* __restrict__ A, int lda,
    const float* __restrict__ W,                 // [N x Kdim], row-major
    float* __restrict__ C, int ldc,
    const float* __restrict__ bias,              // per-n or nullptr
    const float* __restrict__ Cadd,              // residual (ld=ldc) or nullptr
    int M, int N, int Kdim, int act) {           // act: 0 none, 1 softplus
    __shared__ _Float16 As[GBM][GLDST];
    __shared__ _Float16 Ws[GBN][GLDST];
    int tid = threadIdx.x;
    int bm = blockIdx.x * GBM;
    int bn = blockIdx.y * GBN;
    int wave = tid >> 5, lane = tid & 31;
    int wr = wave >> 1, wc = wave & 1;           // 4 x 2 wave grid
    int tm0 = wr * 32, tn0 = wc * 32;
    int lr = lane & 15, hsel = lane >> 4;

    // per-thread staging coordinates (fixed): A -> 4 vec4, W -> 2 vec4
    int ar[4], ac[4];
    #pragma unroll
    for (int ii = 0; ii < 4; ++ii) {
        int i = tid + ii * 256;
        ar[ii] = i >> 3;            // 0..127
        ac[ii] = (i & 7) << 2;      // 0..28
    }
    int wrw[2], wcw[2];
    #pragma unroll
    for (int ii = 0; ii < 2; ++ii) {
        int i = tid + ii * 256;
        wrw[ii] = i >> 3;           // 0..63
        wcw[ii] = (i & 7) << 2;
    }

    f4 ra[4], rw[2];

    auto load_tiles = [&](int k0) {
        #pragma unroll
        for (int ii = 0; ii < 4; ++ii) {
            int gm = bm + ar[ii], gk = k0 + ac[ii];
            f4 v = {0.f, 0.f, 0.f, 0.f};
            if (gm < M && gk < Kdim) v = *(const f4*)(A + (size_t)gm * lda + gk);
            ra[ii] = v;
        }
        #pragma unroll
        for (int ii = 0; ii < 2; ++ii) {
            int gn = bn + wrw[ii], gk = k0 + wcw[ii];
            f4 v = {0.f, 0.f, 0.f, 0.f};
            if (gn < N && gk < Kdim) v = *(const f4*)(W + (size_t)gn * Kdim + gk);
            rw[ii] = v;
        }
    };
    auto store_tiles = [&]() {
        #pragma unroll
        for (int ii = 0; ii < 4; ++ii) {
            h4 h = {(_Float16)ra[ii].x, (_Float16)ra[ii].y,
                    (_Float16)ra[ii].z, (_Float16)ra[ii].w};
            *(h4*)(&As[ar[ii]][ac[ii]]) = h;
        }
        #pragma unroll
        for (int ii = 0; ii < 2; ++ii) {
            h4 h = {(_Float16)rw[ii].x, (_Float16)rw[ii].y,
                    (_Float16)rw[ii].z, (_Float16)rw[ii].w};
            *(h4*)(&Ws[wrw[ii]][wcw[ii]]) = h;
        }
    };

    v8f acc[2][2];
    acc[0][0] = (v8f)0.f; acc[0][1] = (v8f)0.f; acc[1][0] = (v8f)0.f; acc[1][1] = (v8f)0.f;

    union FragU { v16h v; unsigned u[8]; };

    load_tiles(0);
    for (int k0 = 0; k0 < Kdim; k0 += GBK) {
        store_tiles();
        __syncthreads();
        int knext = k0 + GBK;
        if (knext < Kdim) load_tiles(knext);   // issue next tile's loads early

        FragU bf[2];
        #pragma unroll
        for (int j = 0; j < 2; ++j) {
            int n = tn0 + j * 16 + lr;
            #pragma unroll
            for (int e = 0; e < 8; ++e) {
                int kk = (e >> 2) * 16 + hsel * 8 + (e & 3) * 2;
                bf[j].u[e] = *(const unsigned*)&Ws[n][kk];
            }
        }
        #pragma unroll
        for (int i2 = 0; i2 < 2; ++i2) {
            FragU af;
            int m = tm0 + i2 * 16 + lr;
            #pragma unroll
            for (int e = 0; e < 8; ++e) {
                int kk = (e >> 2) * 16 + hsel * 8 + (e & 3) * 2;
                af.u[e] = *(const unsigned*)&As[m][kk];
            }
            #pragma unroll
            for (int j = 0; j < 2; ++j) {
                acc[i2][j] = __builtin_amdgcn_wmma_f32_16x16x32_f16(
                    false, af.v, false, bf[j].v, (short)0, acc[i2][j], false, false);
            }
        }
        __syncthreads();
    }

    #pragma unroll
    for (int i2 = 0; i2 < 2; ++i2) {
        #pragma unroll
        for (int j = 0; j < 2; ++j) {
            int n = bn + tn0 + j * 16 + lr;
            #pragma unroll
            for (int r = 0; r < 8; ++r) {
                int m = bm + tm0 + i2 * 16 + r + 8 * hsel;
                if (m < M && n < N) {
                    float v = acc[i2][j][r];
                    if (bias) v += bias[n];
                    if (act == 1) v = (v > 20.f) ? v : log1pf(__expf(v));
                    if (Cadd) v += Cadd[(size_t)m * ldc + n];
                    C[(size_t)m * ldc + n] = v;
                }
            }
        }
    }
}

// ======================================================================
// Causal depthwise conv (taps=4) over L + SiLU.  xz: [B*L, 2*DI]; use
// first DI columns.  out xc: [B*L, DI].
// ======================================================================
__global__ void conv_silu_kernel(const float* __restrict__ xz,
                                 const float* __restrict__ cw,
                                 const float* __restrict__ cb,
                                 float* __restrict__ xc) {
    int idx = blockIdx.x * blockDim.x + threadIdx.x;
    int total = Bc * Lc * DIc;
    if (idx >= total) return;
    int c = idx % DIc;
    int l = (idx / DIc) % Lc;
    int b = idx / (DIc * Lc);
    const float* base = xz + ((size_t)b * Lc) * (2 * DIc) + c;
    float s = cb[c];
    #pragma unroll
    for (int j = 0; j < DCONVc; ++j) {
        int ll = l - (DCONVc - 1) + j;
        if (ll >= 0) s += base[(size_t)ll * (2 * DIc)] * cw[c * DCONVc + j];
    }
    xc[idx] = silu_f(s);
}

// ======================================================================
// Selective scan: one thread per (b, channel), 16-state recurrence over L.
// x_dbl layout per row: [0:24)=dt(unused here), [24:40)=B, [40:56)=C.
// ======================================================================
__global__ __launch_bounds__(256) void scan_kernel(const float* __restrict__ delta,
                                                   const float* __restrict__ xdbl,
                                                   const float* __restrict__ xc,
                                                   const float* __restrict__ xz,
                                                   const float* __restrict__ Alog,
                                                   const float* __restrict__ Dp,
                                                   float* __restrict__ yact) {
    int idx = blockIdx.x * blockDim.x + threadIdx.x;
    if (idx >= Bc * DIc) return;
    int c = idx % DIc, b = idx / DIc;
    float A[DSc];
    #pragma unroll
    for (int s = 0; s < DSc; ++s) A[s] = -__expf(Alog[(size_t)c * DSc + s]);
    float h[DSc];
    #pragma unroll
    for (int s = 0; s < DSc; ++s) h[s] = 0.f;
    float Dcoef = Dp[c];
    for (int l = 0; l < Lc; ++l) {
        size_t row = (size_t)b * Lc + l;
        float d = delta[row * DIc + c];
        float u = xc[row * DIc + c];
        float z = xz[row * (2 * DIc) + DIc + c];
        const float* Bt = xdbl + row * 56 + DTRc;
        const float* Ct = xdbl + row * 56 + DTRc + DSc;
        float du = d * u;
        float y = 0.f;
        #pragma unroll
        for (int s = 0; s < DSc; ++s) {
            h[s] = h[s] * __expf(d * A[s]) + du * Bt[s];
            y += h[s] * Ct[s];
        }
        y += Dcoef * u;
        y *= silu_f(z);
        yact[row * DIc + c] = y;
    }
}

// ======================================================================
// Head: per-batch mean over tokens + MLP 384->256->64->40 (relu,relu).
// ======================================================================
__global__ __launch_bounds__(256) void head_kernel(const float* __restrict__ tn,
                                                   const float* __restrict__ w1, const float* __restrict__ b1,
                                                   const float* __restrict__ w2, const float* __restrict__ b2,
                                                   const float* __restrict__ w3, const float* __restrict__ b3,
                                                   float* __restrict__ out) {
    int b = blockIdx.x;
    __shared__ float x[Dc];
    __shared__ float h1[256];
    __shared__ float h2[64];
    const float* T = tn + (size_t)b * Lc * Dc;
    int tid = threadIdx.x;
    for (int d = tid; d < Dc; d += 256) {
        float s = 0.f;
        for (int l = 0; l < Lc; ++l) s += T[(size_t)l * Dc + d];
        x[d] = s * (1.0f / (float)Lc);
    }
    __syncthreads();
    {
        float s = b1[tid];
        const float* wr = w1 + (size_t)tid * Dc;
        for (int j = 0; j < Dc; ++j) s += x[j] * wr[j];
        h1[tid] = fmaxf(s, 0.f);
    }
    __syncthreads();
    if (tid < 64) {
        float s = b2[tid];
        const float* wr = w2 + (size_t)tid * 256;
        for (int j = 0; j < 256; ++j) s += h1[j] * wr[j];
        h2[tid] = fmaxf(s, 0.f);
    }
    __syncthreads();
    if (tid < NCc) {
        float s = b3[tid];
        const float* wr = w3 + (size_t)tid * 64;
        for (int j = 0; j < 64; ++j) s += h2[j] * wr[j];
        out[b * NCc + tid] = s;
    }
}

// ======================================================================
extern "C" void kernel_launch(void* const* d_in, const int* in_sizes, int n_in,
                              void* d_out, int out_size, void* d_ws, size_t ws_size,
                              hipStream_t stream) {
    (void)in_sizes; (void)n_in; (void)out_size; (void)ws_size;
    const float* data       = (const float*)d_in[0];
    const float* pe_w1      = (const float*)d_in[1];
    const float* pe_b1      = (const float*)d_in[2];
    const float* pe_g1      = (const float*)d_in[3];
    const float* pe_be1     = (const float*)d_in[4];
    const float* pe_w2      = (const float*)d_in[5];
    const float* pe_b2      = (const float*)d_in[6];
    const float* pe_g2      = (const float*)d_in[7];
    const float* pe_be2     = (const float*)d_in[8];
    const float* pe_w3      = (const float*)d_in[9];
    const float* pe_b3      = (const float*)d_in[10];
    const float* pe_g3      = (const float*)d_in[11];
    const float* pe_be3     = (const float*)d_in[12];
    const float* oi_h_scale = (const float*)d_in[13];
    const float* oi_h_shift = (const float*)d_in[14];
    const float* oi_th_scale= (const float*)d_in[15];
    const float* oi_th_shift= (const float*)d_in[16];
    const float* pos_embed  = (const float*)d_in[17];
    const float* blk_ln_g   = (const float*)d_in[18];
    const float* blk_ln_b   = (const float*)d_in[19];
    const float* blk_in_w   = (const float*)d_in[20];
    const float* blk_conv_w = (const float*)d_in[21];
    const float* blk_conv_b = (const float*)d_in[22];
    const float* blk_xp_w   = (const float*)d_in[23];
    const float* blk_dt_w   = (const float*)d_in[24];
    const float* blk_dt_b   = (const float*)d_in[25];
    const float* blk_Alog   = (const float*)d_in[26];
    const float* blk_D      = (const float*)d_in[27];
    const float* blk_out_w  = (const float*)d_in[28];
    const float* norm_g     = (const float*)d_in[29];
    const float* norm_b     = (const float*)d_in[30];
    const float* mlp_w1     = (const float*)d_in[31];
    const float* mlp_b1     = (const float*)d_in[32];
    const float* mlp_w2     = (const float*)d_in[33];
    const float* mlp_b2     = (const float*)d_in[34];
    const float* mlp_w3     = (const float*)d_in[35];
    const float* mlp_b3     = (const float*)d_in[36];

    // ---- workspace carve-out (all offsets are multiples of 4 floats) ----
    float* ws = (float*)d_ws;
    size_t off = 0;
    float* centers = ws + off; off += (size_t)Bc * Pc * 3;
    float* tokens  = ws + off; off += (size_t)Bc * Pc * Dc;
    float* t       = ws + off; off += (size_t)Bc * Lc * Dc;
    float* xln     = ws + off; off += (size_t)Bc * Lc * Dc;
    float* xz      = ws + off; off += (size_t)Bc * Lc * 2 * DIc;
    float* xc      = ws + off; off += (size_t)Bc * Lc * DIc;
    float* xdbl    = ws + off; off += (size_t)Bc * Lc * 56;
    float* delta   = ws + off; off += (size_t)Bc * Lc * DIc;
    float* yact    = ws + off; off += (size_t)Bc * Lc * DIc;
    float* tn      = ws + off; off += (size_t)Bc * Lc * Dc;
    int* knn = (int*)(ws + off); off += (size_t)Bc * Pc * Kc;
    int* oh  = (int*)(ws + off); off += (size_t)Bc * Pc;
    int* oth = (int*)(ws + off); off += (size_t)Bc * Pc;

    const int Mrows = Bc * Lc;   // 1024

    // ---- patch embed ----
    knn_kernel<<<Bc * Pc, 256, 0, stream>>>(data, centers, knn);
    patch_mlp_kernel<<<Bc * Pc, 256, 0, stream>>>(data, centers, knn,
        pe_w1, pe_b1, pe_g1, pe_be1, pe_w2, pe_b2, pe_g2, pe_be2,
        pe_w3, pe_b3, pe_g3, pe_be3, tokens);
    sfc_kernel<<<Bc, 64, 0, stream>>>(centers, oh, oth);
    {
        int total = Bc * Lc * Dc;
        build_t_kernel<<<(total + 255) / 256, 256, 0, stream>>>(
            tokens, oh, oth, oi_h_scale, oi_h_shift, oi_th_scale, oi_th_shift,
            pos_embed, t);
    }

    // ---- mamba blocks ----
    for (int layer = 0; layer < DEPTHc; ++layer) {
        ln_kernel<<<Mrows, 128, 0, stream>>>(t, blk_ln_g + layer * Dc,
                                             blk_ln_b + layer * Dc, xln, Dc);
        // in_proj: (1024 x 384) x (1536 x 384)^T -> xz (1024 x 1536)
        gemm_wmma_kernel<<<dim3(Mrows / GBM, (2 * DIc) / GBN), 256, 0, stream>>>(
            xln, Dc, blk_in_w + (size_t)layer * 2 * DIc * Dc, xz, 2 * DIc,
            nullptr, nullptr, Mrows, 2 * DIc, Dc, 0);
        {
            int total = Bc * Lc * DIc;
            conv_silu_kernel<<<(total + 255) / 256, 256, 0, stream>>>(
                xz, blk_conv_w + (size_t)layer * DIc * DCONVc,
                blk_conv_b + (size_t)layer * DIc, xc);
        }
        // x_proj: (1024 x 768) x (56 x 768)^T -> xdbl (1024 x 56)
        gemm_wmma_kernel<<<dim3(Mrows / GBM, 1), 256, 0, stream>>>(
            xc, DIc, blk_xp_w + (size_t)layer * 56 * DIc, xdbl, 56,
            nullptr, nullptr, Mrows, 56, DIc, 0);
        // dt_proj: (1024 x 24) x (768 x 24)^T + dt_b -> softplus -> delta
        gemm_wmma_kernel<<<dim3(Mrows / GBM, DIc / GBN), 256, 0, stream>>>(
            xdbl, 56, blk_dt_w + (size_t)layer * DIc * DTRc, delta, DIc,
            blk_dt_b + (size_t)layer * DIc, nullptr, Mrows, DIc, DTRc, 1);
        // selective scan + D skip + z gate
        scan_kernel<<<(Bc * DIc + 255) / 256, 256, 0, stream>>>(
            delta, xdbl, xc, xz,
            blk_Alog + (size_t)layer * DIc * DSc, blk_D + (size_t)layer * DIc,
            yact);
        // out_proj + residual: t += yact @ out_w.T
        gemm_wmma_kernel<<<dim3(Mrows / GBM, Dc / GBN), 256, 0, stream>>>(
            yact, DIc, blk_out_w + (size_t)layer * Dc * DIc, t, Dc,
            nullptr, t, Mrows, Dc, DIc, 0);
    }

    // ---- final norm + head ----
    ln_kernel<<<Mrows, 128, 0, stream>>>(t, norm_g, norm_b, tn, Dc);
    head_kernel<<<Bc, 256, 0, stream>>>(tn, mlp_w1, mlp_b1, mlp_w2, mlp_b2,
                                        mlp_w3, mlp_b3, (float*)d_out);
}
